// SGB_A_39599598469330
// MI455X (gfx1250) — matrix operand order repaired and checked by reading, not compiled, not run on hardware
//
#include <hip/hip_runtime.h>
#include <hip/hip_bf16.h>

// ---------------------------------------------------------------------------
// Types / helpers
// ---------------------------------------------------------------------------
typedef __bf16 v16bf __attribute__((ext_vector_type(16)));
typedef float  v8f   __attribute__((ext_vector_type(8)));
typedef unsigned int u32x4 __attribute__((ext_vector_type(4)));   // POD 128-bit

#define GLOBAL_AS __attribute__((address_space(1)))

union FragU { v16bf v; u32x4 q[2]; };

__device__ __forceinline__ v16bf frag_from2(u32x4 lo, u32x4 hi) {
    FragU f; f.q[0] = lo; f.q[1] = hi; return f.v;
}

__device__ __forceinline__ unsigned short f2bf(float x) {
    union { float f; unsigned int u; } c; c.f = x;
    unsigned int r = c.u + 0x7FFFu + ((c.u >> 16) & 1u);   // RNE
    return (unsigned short)(r >> 16);
}

__device__ __forceinline__ float sigf(float x) { return 1.0f / (1.0f + __expf(-x)); }

// A fragment (16x32 bf16): row-major [16][ldk] in LDS; lane L: M=L&15,
// half=L>>4, elements 0..7 = K (k0+8*half ..), elements 8..15 = K (+16).
__device__ __forceinline__ v16bf loadA(const unsigned short* sh, int ldk, int k0, int lane) {
    int M = lane & 15, half = lane >> 4;
    const unsigned short* row = sh + M * ldk + k0 + 8 * half;
    u32x4 lo = *(const u32x4*)(row);
    u32x4 hi = *(const u32x4*)(row + 16);
    return frag_from2(lo, hi);
}

// B fragment (32x16 bf16): K-major weight Wt[k][n] in GLOBAL memory; lane L
// holds column N=n0+(L&15), half=L>>4, elements 0..7 = K (k0+8*half ..),
// elements 8..15 = K (+16); strided global loads.
__device__ __forceinline__ v16bf loadB(const unsigned short* wt, int ldn, int k0, int n0, int lane) {
    const unsigned short* p = wt + (size_t)(k0 + 8 * (lane >> 4)) * ldn + n0 + (lane & 15);
    const GLOBAL_AS unsigned short* q = (const GLOBAL_AS unsigned short*)p;
    union { v16bf v; unsigned short u[16]; } t;
#pragma unroll
    for (int s = 0; s < 8; ++s) {
        t.u[s]     = q[(size_t)s * ldn];
        t.u[8 + s] = q[(size_t)(16 + s) * ldn];
    }
    return t.v;
}

__device__ __forceinline__ v8f wmma_bf16(v16bf a, v16bf b, v8f c) {
    return __builtin_amdgcn_wmma_f32_16x16x32_bf16(false, a, false, b, (short)0, c, false, false);
}

// Model dims
#define VV 8000
#define DD 256
#define HH 256
#define BB 16
#define TT 128
#define GG 1024        // 4*H
#define NVT 500        // V / 16

// ---------------------------------------------------------------------------
// Weight preparation kernels
// ---------------------------------------------------------------------------
__global__ void k_transpose_bf16(const float* __restrict__ src, unsigned short* __restrict__ dst,
                                 int R, int C) {
    int i = blockIdx.x * 256 + threadIdx.x;
    if (i < R * C) {
        int r = i / C, c = i % C;
        dst[(size_t)c * R + r] = f2bf(src[i]);
    }
}

__global__ void k_cast_bf16(const float* __restrict__ src, unsigned short* __restrict__ dst, int n) {
    int i = blockIdx.x * 256 + threadIdx.x;
    if (i < n) dst[i] = f2bf(src[i]);
}

__global__ void k_bias_sum(const float* __restrict__ a, const float* __restrict__ b,
                           float* __restrict__ o, int n) {
    int i = blockIdx.x * 256 + threadIdx.x;
    if (i < n) o[i] = a[i] + b[i];
}

// ---------------------------------------------------------------------------
// Shared gate-matmul + cell helper. Wave w owns hidden units [32w, 32w+32):
// gate N-tiles {i,f,g,o} x {t2=0,1}. t2-outer keeps only 4 live accumulators;
// A fragment shared across the 4 gate WMMAs; pointers are pre-laundered by
// the caller so B fragments are re-fetched from L2 each step (no scratch
// spills from LICM hoisting).
// ---------------------------------------------------------------------------
__device__ __forceinline__ void lstm_gates_cell(
    const unsigned short* x_sh, const unsigned short* h_sh,
    const unsigned short* pWih, const unsigned short* pWhh, const float* pB,
    int wv, int lane, int lm, float creg[2][8], float hreg[2][8])
{
#pragma unroll
    for (int t2 = 0; t2 < 2; ++t2) {
        v8f acc[4];
#pragma unroll
        for (int g = 0; g < 4; ++g) {
            float bb = pB[256 * g + 32 * wv + 16 * t2 + lm];
#pragma unroll
            for (int r = 0; r < 8; ++r) acc[g][r] = bb;
        }
#pragma unroll
        for (int kt = 0; kt < 8; ++kt) {
            v16bf a = loadA(x_sh, DD, 32 * kt, lane);
#pragma unroll
            for (int g = 0; g < 4; ++g)
                acc[g] = wmma_bf16(a, loadB(pWih, GG, 32 * kt, 256 * g + 32 * wv + 16 * t2, lane), acc[g]);
        }
#pragma unroll
        for (int kt = 0; kt < 8; ++kt) {
            v16bf a = loadA(h_sh, HH, 32 * kt, lane);
#pragma unroll
            for (int g = 0; g < 4; ++g)
                acc[g] = wmma_bf16(a, loadB(pWhh, GG, 32 * kt, 256 * g + 32 * wv + 16 * t2, lane), acc[g]);
        }
        // LSTM cell (gate order i,f,g,o) -> registers only
#pragma unroll
        for (int r = 0; r < 8; ++r) {
            float iv = sigf(acc[0][r]);
            float fv = sigf(acc[1][r]);
            float gv = tanhf(acc[2][r]);
            float ov = sigf(acc[3][r]);
            float c = fv * creg[t2][r] + iv * gv;
            creg[t2][r] = c;
            hreg[t2][r] = ov * tanhf(c);
        }
    }
}

// ---------------------------------------------------------------------------
// Encoder LSTM: grid=2 (dir), block=256 (8 waves). c in registers, h via LDS
// (bf16). henc layout: [dir][t][batch m][unit] fp32.
// ---------------------------------------------------------------------------
__global__ void __launch_bounds__(256) k_encoder(
    const int* __restrict__ sent, const unsigned short* __restrict__ embbf,
    const unsigned short* __restrict__ WihT, const unsigned short* __restrict__ WhhT,
    const float* __restrict__ bsum, float* __restrict__ henc)
{
    const int dir = blockIdx.x;
    const int tid = threadIdx.x, lane = tid & 31, wv = tid >> 5;
    const int half = lane >> 4, lm = lane & 15;

    __shared__ __align__(16) unsigned short x_sh[16 * DD];
    __shared__ __align__(16) unsigned short h_sh[16 * HH];

    for (int i = tid; i < 16 * HH; i += 256) h_sh[i] = 0;   // h0 = 0 (bf16 0)

    float creg[2][8], hreg[2][8];
#pragma unroll
    for (int a = 0; a < 2; ++a)
#pragma unroll
        for (int r = 0; r < 8; ++r) creg[a][r] = 0.0f;

    // x for t=0: <PART>=0 for all rows
    for (int c = tid; c < 16 * 32; c += 256) {
        int row = c >> 5, part = c & 31;
        ((u32x4*)x_sh)[row * 32 + part] = ((const GLOBAL_AS u32x4*)embbf)[part];
    }
    __syncthreads();

    for (int t = 0; t < TT; ++t) {
        // Launder pointers: block cross-iteration CSE/LICM of weight fragments
        const unsigned short* pWih = WihT;
        const unsigned short* pWhh = WhhT;
        const float*          pB   = bsum;
        asm volatile("" : "+s"(pWih), "+s"(pWhh), "+s"(pB));

        lstm_gates_cell(x_sh, h_sh, pWih, pWhh, pB, wv, lane, lm, creg, hreg);
        __syncthreads();   // all reads of x_sh/h_sh done

#pragma unroll
        for (int t2 = 0; t2 < 2; ++t2)
#pragma unroll
            for (int r = 0; r < 8; ++r) {
                int M = r + 8 * half;
                int col = 32 * wv + 16 * t2 + lm;
                h_sh[M * HH + col] = f2bf(hreg[t2][r]);
                henc[(((size_t)dir * TT + t) * 16 + M) * HH + col] = hreg[t2][r];
            }

        // stage x for t+1: tok = vec[:, t] (fwd) / flipped (bwd)
        if (t + 1 < TT) {
            int tn = t + 1;
            for (int c = tid; c < 16 * 32; c += 256) {
                int row = c >> 5, part = c & 31;
                int j = (dir == 0) ? (tn - 1) : (TT - tn);
                int tok = sent[row * TT + j];
                ((u32x4*)x_sh)[row * 32 + part] =
                    ((const GLOBAL_AS u32x4*)(embbf + (size_t)tok * DD))[part];
            }
        }
        __syncthreads();
    }
}

// ---------------------------------------------------------------------------
// Decoder: grid = 2 decoders * 128 row-tiles. Per tile: 4-step LSTM (h0=c0 =
// encoder hidden) fused with V=8000 projection + online logsumexp + gather of
// target-char / <PART> logits. decp layout: [dec][w][n] fp32, n = b*T + t.
// ---------------------------------------------------------------------------
__global__ void __launch_bounds__(256) k_decoder(
    const int* __restrict__ sent, const unsigned short* __restrict__ embbf,
    const unsigned short* __restrict__ WihT_f, const unsigned short* __restrict__ WhhT_f,
    const float* __restrict__ bsum_f, const unsigned short* __restrict__ WpT_f,
    const float* __restrict__ bp_f,
    const unsigned short* __restrict__ WihT_b, const unsigned short* __restrict__ WhhT_b,
    const float* __restrict__ bsum_b, const unsigned short* __restrict__ WpT_b,
    const float* __restrict__ bp_b,
    const float* __restrict__ henc, float* __restrict__ decp)
{
    const int blk  = blockIdx.x;
    const int dec  = blk >> 7;
    const int tile = blk & 127;
    const int n0r  = tile * 16;          // first global row of this tile
    const int b    = n0r >> 7;           // batch index (shared by tile)
    const int t0   = n0r & 127;          // first start-position
    const int tid  = threadIdx.x, lane = tid & 31, wv = tid >> 5;
    const int half = lane >> 4, lm = lane & 15;

    const unsigned short* WihT = dec ? WihT_b : WihT_f;
    const unsigned short* WhhT = dec ? WhhT_b : WhhT_f;
    const unsigned short* WpT  = dec ? WpT_b  : WpT_f;
    const float* bsum = dec ? bsum_b : bsum_f;
    const float* bp   = dec ? bp_b   : bp_f;

    __shared__ __align__(16) unsigned short x_sh[16 * DD];
    __shared__ __align__(16) unsigned short h_sh[16 * HH];
    __shared__ float stats_m[8][16], stats_s[8][16];
    __shared__ float lzbuf[4][16];
    __shared__ float char_raw[3][16];
    __shared__ float tag_raw[4][16];
    __shared__ int   tgt_sh[16];

    // h0 = c0 = encoder hidden at (t0+m, b)
    const float* h0base = henc + (size_t)dec * TT * 16 * HH;
    for (int i = tid; i < 16 * HH; i += 256) {
        int m = i >> 8, u = i & 255;
        h_sh[i] = f2bf(h0base[((size_t)(t0 + m) * 16 + b) * HH + u]);
    }
    float creg[2][8], hreg[2][8];
#pragma unroll
    for (int t2 = 0; t2 < 2; ++t2)
#pragma unroll
        for (int r = 0; r < 8; ++r) {
            int M = r + 8 * half;
            int col = 32 * wv + 16 * t2 + lm;
            creg[t2][r] = h0base[((size_t)(t0 + M) * 16 + b) * HH + col];
        }
    // x for step 0: <PART>
    for (int c = tid; c < 16 * 32; c += 256) {
        int row = c >> 5, part = c & 31;
        ((u32x4*)x_sh)[row * 32 + part] = ((const GLOBAL_AS u32x4*)embbf)[part];
    }
    __syncthreads();

    for (int s = 0; s < 4; ++s) {
        // Launder pointers: block cross-iteration CSE/LICM of weight fragments
        const unsigned short* pWih = WihT;
        const unsigned short* pWhh = WhhT;
        const unsigned short* pWp  = WpT;
        const float*          pB   = bsum;
        const float*          pbp  = bp;
        asm volatile("" : "+s"(pWih), "+s"(pWhh), "+s"(pWp), "+s"(pB), "+s"(pbp));

        // ---- LSTM gates + cell (registers only) ----
        lstm_gates_cell(x_sh, h_sh, pWih, pWhh, pB, wv, lane, lm, creg, hreg);
        __syncthreads();   // all reads of x_sh/h_sh done

        // ---- publish h ----
#pragma unroll
        for (int t2 = 0; t2 < 2; ++t2)
#pragma unroll
            for (int r = 0; r < 8; ++r) {
                int M = r + 8 * half;
                int col = 32 * wv + 16 * t2 + lm;
                h_sh[M * HH + col] = f2bf(hreg[t2][r]);
            }

        // ---- stage x for step s+1 and target char col for this step ----
        if (s < 3) {
            // token j = min(t0+row+s, T-1); vec is flipped for backward decoder
            for (int c = tid; c < 16 * 32; c += 256) {
                int row = c >> 5, part = c & 31;
                int j = t0 + row + s; if (j > TT - 1) j = TT - 1;
                int tok = (dec == 0) ? sent[b * TT + j] : sent[b * TT + (TT - 1 - j)];
                ((u32x4*)x_sh)[row * 32 + part] =
                    ((const GLOBAL_AS u32x4*)(embbf + (size_t)tok * DD))[part];
            }
            if (tid < 16) {
                int j = t0 + tid + s; if (j > TT - 1) j = TT - 1;
                tgt_sh[tid] = (dec == 0) ? sent[b * TT + j] : sent[b * TT + (TT - 1 - j)];
            }
        }
        __syncthreads();

        // ---- fused projection + online logsumexp + gather ----
        float lm8[8], ls8[8];
#pragma unroll
        for (int r = 0; r < 8; ++r) { lm8[r] = -1e30f; ls8[r] = 0.0f; }

        for (int vt = wv; vt < NVT; vt += 8) {
            int nv0 = vt * 16;
            int cb = nv0 + lm;
            float bb = pbp[cb];
            v8f a;
#pragma unroll
            for (int r = 0; r < 8; ++r) a[r] = bb;
#pragma unroll
            for (int kt = 0; kt < 8; ++kt)
                a = wmma_bf16(loadA(h_sh, HH, 32 * kt, lane), loadB(pWp, VV, 32 * kt, nv0, lane), a);
#pragma unroll
            for (int r = 0; r < 8; ++r) {
                float v = a[r];
                if (v > lm8[r]) { ls8[r] = ls8[r] * __expf(lm8[r] - v) + 1.0f; lm8[r] = v; }
                else            { ls8[r] += __expf(v - lm8[r]); }
                int M = r + 8 * half;
                if (s < 3 && cb == tgt_sh[M]) char_raw[s][M] = v;
                if (s >= 1 && cb == 0)        tag_raw[s][M]  = v;   // <PART>=0
            }
        }
        // reduce (m,s) across the 16 lanes of each half
        for (int d = 1; d < 16; d <<= 1) {
#pragma unroll
            for (int r = 0; r < 8; ++r) {
                float mo = __shfl_xor(lm8[r], d, 32);
                float so = __shfl_xor(ls8[r], d, 32);
                float mm = fmaxf(lm8[r], mo);
                ls8[r] = ls8[r] * __expf(lm8[r] - mm) + so * __expf(mo - mm);
                lm8[r] = mm;
            }
        }
        if (lm == 0) {
#pragma unroll
            for (int r = 0; r < 8; ++r) {
                int M = r + 8 * half;
                stats_m[wv][M] = lm8[r];
                stats_s[wv][M] = ls8[r];
            }
        }
        __syncthreads();
        if (tid < 16) {
            float mm = -1e30f;
#pragma unroll
            for (int w = 0; w < 8; ++w) mm = fmaxf(mm, stats_m[w][tid]);
            float ss = 0.0f;
#pragma unroll
            for (int w = 0; w < 8; ++w) ss += stats_s[w][tid] * __expf(stats_m[w][tid] - mm);
            lzbuf[s][tid] = mm + logf(ss);
        }
        __syncthreads();
    }

    // p[w][m] = cumsum(char log-probs) + tag log-prob
    if (tid < 16) {
        int m = tid;
        float cum = 0.0f;
#pragma unroll
        for (int w = 0; w < 3; ++w) {
            cum += char_raw[w][m] - lzbuf[w][m];
            float p = cum + (tag_raw[w + 1][m] - lzbuf[w + 1][m]);
            decp[((size_t)dec * 3 + w) * (BB * TT) + n0r + m] = p;
        }
    }
}

// ---------------------------------------------------------------------------
// Semi-CRF forward recursion + loss. 1 block, 1 wave; lane b < 16 handles one
// batch column. fwdP[t][w][b] = decp[0][w][b*T+t]; bwdP likewise with dec=1.
// ---------------------------------------------------------------------------
__global__ void k_crf(const float* __restrict__ decp, float* __restrict__ out) {
    int bl = threadIdx.x;
    const float* pf = decp;
    const float* pb = decp + 3 * (BB * TT);
    float buf0 = 0.0f, buf1 = 0.0f, buf2 = 0.0f, tot = 0.0f;
    if (bl < 16) {
        for (int j = 1; j <= TT; ++j) {
            int x = j - 1;
            float pv[3];
#pragma unroll
            for (int i = 0; i < 3; ++i) {
                int tf = x - i; if (tf < 0) tf = 0;
                float fv = pf[i * (BB * TT) + bl * TT + tf];
                float bv = pb[i * (BB * TT) + bl * TT + (TT - 1 - x)];
                pv[i] = 0.5f * (fv + bv);
            }
            float c0 = buf0 + pv[0];
            float c1 = (j > 1) ? (buf1 + pv[1]) : -1e30f;
            float c2 = (j > 2) ? (buf2 + pv[2]) : -1e30f;
            float mm = fmaxf(c0, fmaxf(c1, c2));
            tot = mm + logf(__expf(c0 - mm) + __expf(c1 - mm) + __expf(c2 - mm));
            buf2 = buf1; buf1 = buf0; buf0 = tot;
        }
    }
    float v = (bl < 16) ? tot : 0.0f;
#pragma unroll
    for (int d = 1; d < 16; d <<= 1) v += __shfl_xor(v, d, 32);
    if (threadIdx.x == 0) out[0] = -v / 16.0f;
}

// ---------------------------------------------------------------------------
// Host launcher
// ---------------------------------------------------------------------------
extern "C" void kernel_launch(void* const* d_in, const int* in_sizes, int n_in,
                              void* d_out, int out_size, void* d_ws, size_t ws_size,
                              hipStream_t stream) {
    (void)in_sizes; (void)n_in; (void)out_size;

    const int*   sent     = (const int*)d_in[0];
    const float* emb      = (const float*)d_in[1];
    const float* eWih     = (const float*)d_in[2];
    const float* eWhh     = (const float*)d_in[3];
    const float* eBih     = (const float*)d_in[4];
    const float* eBhh     = (const float*)d_in[5];
    const float* fWih     = (const float*)d_in[6];
    const float* fWhh     = (const float*)d_in[7];
    const float* fBih     = (const float*)d_in[8];
    const float* fBhh     = (const float*)d_in[9];
    const float* fWp      = (const float*)d_in[10];
    const float* fBp      = (const float*)d_in[11];
    const float* bWih     = (const float*)d_in[12];
    const float* bWhh     = (const float*)d_in[13];
    const float* bBih     = (const float*)d_in[14];
    const float* bBhh     = (const float*)d_in[15];
    const float* bWp      = (const float*)d_in[16];
    const float* bBp      = (const float*)d_in[17];

    char* ws = (char*)d_ws;
    size_t off = 0;
    auto take = [&](size_t bytes) -> char* {
        char* p = ws + off;
        off += (bytes + 255) & ~(size_t)255;
        return p;
    };

    unsigned short* encWihT = (unsigned short*)take((size_t)GG * DD * 2);   // [256][1024]
    unsigned short* encWhhT = (unsigned short*)take((size_t)GG * HH * 2);
    unsigned short* dfWihT  = (unsigned short*)take((size_t)GG * DD * 2);
    unsigned short* dfWhhT  = (unsigned short*)take((size_t)GG * HH * 2);
    unsigned short* dbWihT  = (unsigned short*)take((size_t)GG * DD * 2);
    unsigned short* dbWhhT  = (unsigned short*)take((size_t)GG * HH * 2);
    unsigned short* dfWpT   = (unsigned short*)take((size_t)VV * HH * 2);   // [256][8000]
    unsigned short* dbWpT   = (unsigned short*)take((size_t)VV * HH * 2);
    unsigned short* embbf   = (unsigned short*)take((size_t)VV * DD * 2);
    float* encB = (float*)take(GG * 4);
    float* dfB  = (float*)take(GG * 4);
    float* dbB  = (float*)take(GG * 4);
    float* henc = (float*)take((size_t)2 * TT * 16 * HH * 4);
    float* decp = (float*)take((size_t)2 * 3 * BB * TT * 4);
    (void)ws_size;

    const int nLstm = GG * DD;           // 1024*256
    const int nProj = VV * HH;           // 8000*256
    dim3 b256(256);
    dim3 gL((nLstm + 255) / 256), gP((nProj + 255) / 256), gB((GG + 255) / 256);

    k_transpose_bf16<<<gL, b256, 0, stream>>>(eWih, encWihT, GG, DD);
    k_transpose_bf16<<<gL, b256, 0, stream>>>(eWhh, encWhhT, GG, HH);
    k_transpose_bf16<<<gL, b256, 0, stream>>>(fWih, dfWihT, GG, DD);
    k_transpose_bf16<<<gL, b256, 0, stream>>>(fWhh, dfWhhT, GG, HH);
    k_transpose_bf16<<<gL, b256, 0, stream>>>(bWih, dbWihT, GG, DD);
    k_transpose_bf16<<<gL, b256, 0, stream>>>(bWhh, dbWhhT, GG, HH);
    k_transpose_bf16<<<gP, b256, 0, stream>>>(fWp, dfWpT, VV, HH);
    k_transpose_bf16<<<gP, b256, 0, stream>>>(bWp, dbWpT, VV, HH);
    k_cast_bf16<<<gP, b256, 0, stream>>>(emb, embbf, nProj);
    k_bias_sum<<<gB, b256, 0, stream>>>(eBih, eBhh, encB, GG);
    k_bias_sum<<<gB, b256, 0, stream>>>(fBih, fBhh, dfB, GG);
    k_bias_sum<<<gB, b256, 0, stream>>>(bBih, bBhh, dbB, GG);

    k_encoder<<<2, 256, 0, stream>>>(sent, embbf, encWihT, encWhhT, encB, henc);

    k_decoder<<<256, 256, 0, stream>>>(sent, embbf,
                                       dfWihT, dfWhhT, dfB, dfWpT, fBp,
                                       dbWihT, dbWhhT, dbB, dbWpT, bBp,
                                       henc, decp);

    k_crf<<<1, 32, 0, stream>>>(decp, (float*)d_out);
}
